// RewireGearnet_63024350102002
// MI455X (gfx1250) — matrix operand order repaired and checked by reading, not compiled, not run on hardware
//
#include <hip/hip_runtime.h>
#include <hip/hip_bf16.h>

typedef __attribute__((ext_vector_type(8)))  _Float16 v8h;
typedef __attribute__((ext_vector_type(16))) _Float16 v16h;
typedef __attribute__((ext_vector_type(4)))  _Float16 v4h;
typedef __attribute__((ext_vector_type(8)))  float    v8f;

#define DD    128            // feature dim / out_dim
#define NREL  7
#define NB    8              // 7 relation blocks + 1 self block
#define YW    (NB * DD)      // 1024, y row width

// ---- P1: cast x -> f16 ---------------------------------------------------
__global__ void RG_cast_x_kernel(const float* __restrict__ x,
                                 _Float16* __restrict__ xh, long long n) {
    long long i = (long long)blockIdx.x * blockDim.x + threadIdx.x;
    if (i < n) xh[i] = (_Float16)x[i];
}

// ---- P2: build transposed f16 weight basis Wt[j][k] = Bcat[k][j] ---------
// Bcat[k][rel*128+c] = W_lin[(rel*128+k)*128 + c];  Bcat[k][896+c] = W_self[k*128+c]
__global__ void RG_wt_kernel(const float* __restrict__ Wlin,
                             const float* __restrict__ Wself,
                             _Float16* __restrict__ Wt) {
    int idx = blockIdx.x * blockDim.x + threadIdx.x;   // 1024*128 tasks
    if (idx >= YW * DD) return;
    int j = idx >> 7;          // output column 0..1023
    int k = idx & 127;         // K index 0..127
    int c = j & 127;
    float v = (j < NREL * DD) ? Wlin[((j >> 7) * DD + k) * DD + c]
                              : Wself[k * DD + c];
    Wt[(long long)j * DD + k] = (_Float16)v;
}

// ---- K1: y[N,1024] = f16( x @ [Wlin blocks | Wself] ), WMMA f16 ----------
// No LDS: fragments loaded directly from global, lane half h holding k = 16h..16h+15 (contiguous).
__global__ __launch_bounds__(256)
void RG_gemm_kernel(const _Float16* __restrict__ xh,   // [N,128]
                    const _Float16* __restrict__ Wt,   // [1024,128] transposed basis
                    _Float16*       __restrict__ y,    // [N,1024]
                    int N) {
    const int lane  = threadIdx.x & 31;
    const int wave  = threadIdx.x >> 5;               // 0..7
    const int mbase = blockIdx.x * 16;
    const int cg    = blockIdx.y * DD + wave * 16;    // this wave's 16-col tile
    const int col   = cg + (lane & 15);
    const int half  = lane >> 4;                      // 0 or 1
    const int khA   = half * 16;                      // A frag K sub-offset
    const int khB   = half * 16;                      // B frag K sub-offset

    int arow = mbase + (lane & 15); if (arow >= N) arow = N - 1;
    const _Float16* aptr = xh + (long long)arow * DD;
    const _Float16* bptr = Wt + (long long)col * DD;

    v8f c = {};
    #pragma unroll
    for (int kk = 0; kk < DD; kk += 32) {
        // A 16x32 f16 frag: lane(0-15): K=kk+0..15 ; lane(16-31): K=kk+16..31
        v8h a0 = *(const v8h*)(aptr + kk + khA);
        v8h a1 = *(const v8h*)(aptr + kk + 8 + khA);
        v16h a = __builtin_shufflevector(a0, a1, 0,1,2,3,4,5,6,7,
                                                 8,9,10,11,12,13,14,15);
        // B 32x16 f16 frag: lane(0-15): K=kk+0..15 ; lane(16-31): K=kk+16..31
        v16h b = *(const v16h*)(bptr + kk + khB);
        c = __builtin_amdgcn_wmma_f32_16x16x32_f16(
                false, a, false, b, (short)0, c, false, false);
    }

    // D layout: c[i] -> rows (i, i+8) per half-wave, col = cg + (lane&15)
    _Float16* yout = y + (long long)(mbase + half * 8) * YW + col;
    if (mbase + 16 <= N) {               // block-uniform fast path (no exec juggling)
        #pragma unroll
        for (int i = 0; i < 8; ++i)
            yout[(long long)i * YW] = (_Float16)c[i];
    } else {
        #pragma unroll
        for (int i = 0; i < 8; ++i)
            if (mbase + half * 8 + i < N)
                yout[(long long)i * YW] = (_Float16)c[i];
    }
}

// ---- Z: zero the 51MB accumulator ---------------------------------------
__global__ void RG_zero_kernel(float4* __restrict__ p, long long n4) {
    long long i = (long long)blockIdx.x * blockDim.x + threadIdx.x;
    if (i < n4) p[i] = make_float4(0.f, 0.f, 0.f, 0.f);
}

// ---- S: acc[dst] += w_e * y[src, rel]  (f32 atomics, L2-resident dest) ---
// One wave per edge: edge metadata is wave-uniform -> scalar (SMEM) loads.
__global__ __launch_bounds__(256)
void RG_scatter_kernel(const _Float16* __restrict__ y,
                       const int*   __restrict__ ei,   // [2,E]
                       const int*   __restrict__ rel,
                       const float* __restrict__ ew,
                       float*       __restrict__ acc,  // [N,128]
                       int E) {
    int e = blockIdx.x * 8 + (threadIdx.x >> 5);   // wave-uniform edge id
    if (e >= E) return;
    int j = (threadIdx.x & 31) * 4;                // lane's 4 columns
    int   src = ei[e];
    int   dst = ei[(long long)E + e];
    int   r   = rel[e];
    float w   = ew[e];
    v4h yv = *(const v4h*)(y + (long long)src * YW + r * DD + j);
    float* base = acc + (long long)dst * DD + j;
    unsafeAtomicAdd(base + 0, w * (float)yv[0]);
    unsafeAtomicAdd(base + 1, w * (float)yv[1]);
    unsafeAtomicAdd(base + 2, w * (float)yv[2]);
    unsafeAtomicAdd(base + 3, w * (float)yv[3]);
}

// ---- E: out = relu(acc + y_self + b_lin + b_self) ------------------------
__global__ void RG_epilogue_kernel(const float* __restrict__ acc,
                                   const _Float16* __restrict__ y,
                                   const float* __restrict__ blin,
                                   const float* __restrict__ bself,
                                   float* __restrict__ out, long long n) {
    long long i = (long long)blockIdx.x * blockDim.x + threadIdx.x;
    if (i >= n) return;
    long long node = i >> 7;
    int c = (int)i & 127;
    float v = acc[i] + (float)y[node * YW + NREL * DD + c] + blin[c] + bself[c];
    out[i] = v > 0.f ? v : 0.f;
}

extern "C" void kernel_launch(void* const* d_in, const int* in_sizes, int n_in,
                              void* d_out, int out_size, void* d_ws, size_t ws_size,
                              hipStream_t stream) {
    const float* x     = (const float*)d_in[0];
    const int*   ei    = (const int*)  d_in[1];
    const int*   rel   = (const int*)  d_in[2];
    const float* ew    = (const float*)d_in[3];
    const float* Wlin  = (const float*)d_in[4];
    const float* blin  = (const float*)d_in[5];
    const float* Wself = (const float*)d_in[6];
    const float* bself = (const float*)d_in[7];
    float* out = (float*)d_out;

    long long N = in_sizes[0] / DD;
    int E = in_sizes[3];

    // workspace layout (bytes):
    //   y   : N*1024 f16   (204.8 MB)
    //   acc : N*128  f32   ( 51.2 MB)
    //   xh  : N*128  f16   ( 25.6 MB)
    //   Wt  : 1024*128 f16 (  0.25 MB)
    char* ws = (char*)d_ws;
    _Float16* yh  = (_Float16*)ws;
    float*    acc = (float*)(ws + N * YW * 2);
    _Float16* xh  = (_Float16*)(ws + N * YW * 2 + N * DD * 4);
    _Float16* Wt  = (_Float16*)(ws + N * YW * 2 + N * DD * 4 + N * DD * 2);

    long long nx = N * DD;                                   // 12.8M
    RG_cast_x_kernel<<<(int)((nx + 255) / 256), 256, 0, stream>>>(x, xh, nx);

    RG_wt_kernel<<<(YW * DD + 255) / 256, 256, 0, stream>>>(Wlin, Wself, Wt);

    dim3 ggrid((unsigned)((N + 15) / 16), NB);
    RG_gemm_kernel<<<ggrid, 256, 0, stream>>>(xh, Wt, yh, (int)N);

    long long n4 = N * DD / 4;
    RG_zero_kernel<<<(int)((n4 + 255) / 256), 256, 0, stream>>>((float4*)acc, n4);

    int sb = (E + 7) / 8;    // 8 waves (edges) per 256-thread block
    RG_scatter_kernel<<<sb, 256, 0, stream>>>(yh, ei, rel, ew, acc, E);

    RG_epilogue_kernel<<<(int)((nx + 255) / 256), 256, 0, stream>>>(
        acc, yh, blin, bself, out, nx);
}
